// MAML_83313775608374
// MI455X (gfx1250) — matrix operand order repaired and checked by reading, not compiled, not run on hardware
//
#include <hip/hip_runtime.h>
#include <math.h>

// MI455X / gfx1250, wave32. WMMA f32 <= f16 16x16x32, f16-staged operands.

typedef __attribute__((ext_vector_type(16))) _Float16 v16h;
typedef __attribute__((ext_vector_type(8)))  _Float16 v8h;
typedef __attribute__((ext_vector_type(8)))  float    v8f;

#define SEQ 128
#define DIM 256
#define NH  4
#define DH  64
#define FFD 384
#define NL  6

__device__ __forceinline__ v8f wmma_f16(v16h a, v16h b, v8f c) {
  // (neg_a, A, neg_b, B, c_mod, C, reuse_a, reuse_b)
  return __builtin_amdgcn_wmma_f32_16x16x32_f16(false, a, false, b, (short)0, c,
                                                false, false);
}

// A fragment 16x32 (MxK) from f16 row-major [*, lda].
// lane<16:  row=mb+lane,    elems 0..15 -> K kb..kb+15
// lane>=16: row=mb+lane-16, elems 0..15 -> K kb+16..kb+31
__device__ __forceinline__ v16h load_a_frag16(const _Float16* A, int lda, int mb,
                                              int kb, int lane) {
  int m  = mb + (lane & 15);
  int hi = (lane >> 4) & 1;
  const _Float16* p = A + (long)m * lda + kb + hi * 16;
  v8h lo  = *(const v8h*)p;         // 16B load
  v8h hi8 = *(const v8h*)(p + 8);   // 16B load
  return __builtin_shufflevector(lo, hi8, 0, 1, 2, 3, 4, 5, 6, 7,
                                 8, 9, 10, 11, 12, 13, 14, 15);
}

// B fragment 32x16 (KxN) for C = A*W^T, W f16 row-major [N,K].
// lane<16: col n=nb+lane holds K kb..kb+15 ; lane>=16: col n=nb+lane-16, K kb+16..kb+31
__device__ __forceinline__ v16h load_bt_frag16(const _Float16* W, int ldw, int nb,
                                               int kb, int lane) {
  int n  = nb + (lane & 15);
  int hi = (lane >> 4) & 1;
  return *(const v16h*)(W + (long)n * ldw + kb + hi * 16);  // 32B load
}

// B fragment for non-transposed f16 B stored [K,N] row-major (used for P*V).
__device__ __forceinline__ v16h load_b_frag16(const _Float16* B, int ldb, int nb,
                                              int kb, int lane) {
  int n  = nb + (lane & 15);
  int hi = (lane >> 4) & 1;
  const _Float16* p = B + (long)(kb + hi * 16) * ldb + n;
  v16h b;
#pragma unroll
  for (int e = 0; e < 16; ++e) b[e] = p[(long)e * ldb];
  return b;
}

// ---------------- f32 -> f16 bulk convert (weights, once per launch) ---------
__global__ void cvt16_kernel(const float* __restrict__ in,
                             _Float16* __restrict__ out, long n) {
  long i = (long)blockIdx.x * blockDim.x + threadIdx.x;
  if (i < n) out[i] = (_Float16)in[i];
}

// ---------------- embedding + positional encoding ----------------
__global__ void embed_kernel(const int* __restrict__ x,
                             const float* __restrict__ E,
                             float* __restrict__ h, long total) {
  long idx = (long)blockIdx.x * blockDim.x + threadIdx.x;
  if (idx >= total) return;
  int  d = (int)(idx % DIM);
  long m = idx / DIM;
  int  s = (int)(m % SEQ);
  int  i = d >> 1;
  float dim_t = powf(10000.0f, (float)(2 * i) / (float)DIM);
  float ang   = (float)s / dim_t;
  float pe    = (d & 1) ? cosf(ang) : sinf(ang);
  h[idx] = E[(long)x[m] * DIM + d] + pe;
}

// ---------- layernorm: one wave32 per row of 256; emits f32 + f16 ----------
__global__ void __launch_bounds__(256) layernorm_kernel(
    const float* __restrict__ X, const float* __restrict__ w,
    const float* __restrict__ bb, float* __restrict__ Y,
    _Float16* __restrict__ Y16, int M) {
  int lane = threadIdx.x & 31;
  int wave = threadIdx.x >> 5;
  int row  = blockIdx.x * 8 + wave;
  if (row >= M) return;
  const float* xr = X + (long)row * DIM;
  float v[8];
  float s = 0.f;
#pragma unroll
  for (int j = 0; j < 8; ++j) { v[j] = xr[lane + j * 32]; s += v[j]; }
#pragma unroll
  for (int off = 16; off > 0; off >>= 1) s += __shfl_xor(s, off, 32);
  float mu  = s * (1.0f / 256.0f);
  float var = 0.f;
#pragma unroll
  for (int j = 0; j < 8; ++j) { float d = v[j] - mu; var += d * d; }
#pragma unroll
  for (int off = 16; off > 0; off >>= 1) var += __shfl_xor(var, off, 32);
  float inv = rsqrtf(var * (1.0f / 256.0f) + 1e-5f);
  float*     yr  = Y + (long)row * DIM;
  _Float16*  yr6 = Y16 + (long)row * DIM;
#pragma unroll
  for (int j = 0; j < 8; ++j) {
    int   d = lane + j * 32;
    float y = (v[j] - mu) * inv * w[d] + bb[d];
    yr[d]  = y;
    yr6[d] = (_Float16)y;
  }
}

// -------- WMMA GEMM, 32x64 per wave: C = A16 * W16^T (+bias)(gelu)(+res) -----
template <int HAS_BIAS, int DO_GELU, int HAS_RES, int OUT_F16>
__global__ void __launch_bounds__(256) gemm16_kernel(
    const _Float16* __restrict__ A, const _Float16* __restrict__ W,
    const float* __restrict__ bias, const float* __restrict__ Res,
    float* __restrict__ C, _Float16* __restrict__ C16, int M, int N, int K) {
  int lane = threadIdx.x & 31;
  int wave = threadIdx.x >> 5;
  int tn   = N >> 6;                      // 64-wide N strips
  int tile = blockIdx.x * 8 + wave;       // wave-uniform
  if (tile >= (M >> 5) * tn) return;      // wave-uniform exit, EXEC stays full
  int mb = (tile / tn) << 5;
  int nb = (tile % tn) << 6;
  v8f acc[2][4] = {};
  for (int kb = 0; kb < K; kb += 32) {
    if (kb + 32 < K) {
      __builtin_prefetch(A + (long)(mb + (lane & 15)) * K + kb + 32);
      __builtin_prefetch(W + (long)(nb + (lane & 15)) * K + kb + 32);
    }
    v16h a0 = load_a_frag16(A, K, mb, kb, lane);
    v16h a1 = load_a_frag16(A, K, mb + 16, kb, lane);
    v16h b0 = load_bt_frag16(W, K, nb, kb, lane);
    v16h b1 = load_bt_frag16(W, K, nb + 16, kb, lane);
    v16h b2 = load_bt_frag16(W, K, nb + 32, kb, lane);
    v16h b3 = load_bt_frag16(W, K, nb + 48, kb, lane);
    acc[0][0] = wmma_f16(a0, b0, acc[0][0]);
    acc[0][1] = wmma_f16(a0, b1, acc[0][1]);
    acc[0][2] = wmma_f16(a0, b2, acc[0][2]);
    acc[0][3] = wmma_f16(a0, b3, acc[0][3]);
    acc[1][0] = wmma_f16(a1, b0, acc[1][0]);
    acc[1][1] = wmma_f16(a1, b1, acc[1][1]);
    acc[1][2] = wmma_f16(a1, b2, acc[1][2]);
    acc[1][3] = wmma_f16(a1, b3, acc[1][3]);
  }
  int hi = (lane >> 4) & 1;
  int nl = lane & 15;
#pragma unroll
  for (int mi = 0; mi < 2; ++mi) {
#pragma unroll
    for (int nj = 0; nj < 4; ++nj) {
      int   n  = nb + nj * 16 + nl;
      float bv = HAS_BIAS ? bias[n] : 0.0f;
#pragma unroll
      for (int i = 0; i < 8; ++i) {
        int   m = mb + mi * 16 + hi * 8 + i;  // C layout: VGPR i -> M=i / M=8+i
        float c = acc[mi][nj][i] + bv;
        if (DO_GELU) c = 0.5f * c * (1.0f + erff(c * 0.70710678118654752f));
        if (HAS_RES) c += Res[(long)m * N + n];
        if (OUT_F16) C16[(long)m * N + n] = (_Float16)c;
        else         C[(long)m * N + n]   = c;
      }
    }
  }
}

// ---------------- fused attention per (batch, head), f16 QKV -> f16 O --------
// qkv16: [B*S, 768] f16; q at col h*64, k at 256+h*64, v at 512+h*64
__global__ void __launch_bounds__(256) attn_kernel(
    const _Float16* __restrict__ qkv, const float* __restrict__ att_ids,
    _Float16* __restrict__ o) {
  __shared__ __align__(16) float    sc[SEQ * SEQ];  // 64 KB scores (f32)
  __shared__ __align__(16) _Float16 ps[SEQ * SEQ];  // 32 KB probabilities (f16)
  int b    = blockIdx.x >> 2;
  int hh   = blockIdx.x & 3;
  int lane = threadIdx.x & 31;
  int wave = threadIdx.x >> 5;
  const _Float16* base = qkv + (long)b * SEQ * (3 * DIM);
  const _Float16* Q    = base + hh * DH;
  const _Float16* Km   = base + DIM + hh * DH;
  const _Float16* V    = base + 2 * DIM + hh * DH;

  // Phase 1: S = Q * K^T  (8x8 tiles of 16x16, K-dim = 64)
  for (int t = wave; t < 64; t += 8) {
    int mt = (t >> 3) << 4, nt = (t & 7) << 4;
    v8f acc = {};
    for (int kb = 0; kb < DH; kb += 32) {
      v16h a  = load_a_frag16(Q, 3 * DIM, mt, kb, lane);
      v16h bf = load_bt_frag16(Km, 3 * DIM, nt, kb, lane);
      acc = wmma_f16(a, bf, acc);
    }
    int n  = nt + (lane & 15);
    int hi = (lane >> 4) & 1;
#pragma unroll
    for (int i = 0; i < 8; ++i) sc[(mt + hi * 8 + i) * SEQ + n] = acc[i];
  }
  __syncthreads();

  // Phase 2: masked, scaled softmax per row; f16 probs into ps
  if (threadIdx.x < SEQ) {
    int   r  = threadIdx.x;
    float mx = -3.4e38f;
    for (int kk = 0; kk < SEQ; ++kk) {
      float val = (sc[r * SEQ + kk] + att_ids[b * SEQ + kk] * -1e9f) * 0.125f;
      sc[r * SEQ + kk] = val;
      mx = fmaxf(mx, val);
    }
    float sum = 0.f;
    for (int kk = 0; kk < SEQ; ++kk) {
      float e = __expf(sc[r * SEQ + kk] - mx);
      sc[r * SEQ + kk] = e;
      sum += e;
    }
    float inv = 1.0f / sum;
    for (int kk = 0; kk < SEQ; ++kk)
      ps[r * SEQ + kk] = (_Float16)(sc[r * SEQ + kk] * inv);
  }
  __syncthreads();

  // Phase 3: O = P * V  (8x4 tiles, K-dim = 128), P from LDS via ds_load_b128
  for (int t = wave; t < 32; t += 8) {
    int mt = (t >> 2) << 4, nt = (t & 3) << 4;
    v8f acc = {};
    for (int kb = 0; kb < SEQ; kb += 32) {
      v16h a  = load_a_frag16(ps, SEQ, mt, kb, lane);
      v16h bf = load_b_frag16(V, 3 * DIM, nt, kb, lane);
      acc = wmma_f16(a, bf, acc);
    }
    int n  = nt + (lane & 15);
    int hi = (lane >> 4) & 1;
#pragma unroll
    for (int i = 0; i < 8; ++i)
      o[((long)b * SEQ + mt + hi * 8 + i) * DIM + hh * DH + n] =
          (_Float16)acc[i];
  }
}

// ---------------- final classifier: out[b,c] = h_flat . W[c] + bias ----------
__global__ void __launch_bounds__(256) logits_kernel(
    const float* __restrict__ hflat, const float* __restrict__ W,
    const float* __restrict__ bias, float* __restrict__ out) {
  int b = blockIdx.x;
  int t = threadIdx.x;
  const float* x = hflat + (long)b * (SEQ * DIM);
  float s0 = 0.f, s1 = 0.f;
  for (int i = t; i < SEQ * DIM; i += 256) {
    float xv = x[i];
    s0 += xv * W[i];
    s1 += xv * W[SEQ * DIM + i];
  }
  __shared__ float r0[256], r1[256];
  r0[t] = s0; r1[t] = s1;
  __syncthreads();
  for (int off = 128; off > 0; off >>= 1) {
    if (t < off) { r0[t] += r0[t + off]; r1[t] += r1[t + off]; }
    __syncthreads();
  }
  if (t == 0) {
    out[b * 2 + 0] = r0[0] + bias[0];
    out[b * 2 + 1] = r1[0] + bias[1];
  }
}

extern "C" void kernel_launch(void* const* d_in, const int* in_sizes, int n_in,
                              void* d_out, int out_size, void* d_ws,
                              size_t ws_size, hipStream_t stream) {
  const int M  = in_sizes[0];      // B*S = 65536
  const int Bn = M / SEQ;          // 512

  const int*   x          = (const int*)d_in[0];
  const float* att        = (const float*)d_in[1];
  const float* embed_w    = (const float*)d_in[2];
  const float* norm1_w    = (const float*)d_in[3];
  const float* norm1_b    = (const float*)d_in[4];
  const float* proj_in_w  = (const float*)d_in[5];
  const float* proj_out_w = (const float*)d_in[6];
  const float* proj_out_b = (const float*)d_in[7];
  const float* norm2_w    = (const float*)d_in[8];
  const float* norm2_b    = (const float*)d_in[9];
  const float* lin1_w     = (const float*)d_in[10];
  const float* lin1_b     = (const float*)d_in[11];
  const float* lin2_w     = (const float*)d_in[12];
  const float* lin2_b     = (const float*)d_in[13];
  const float* logits_w   = (const float*)d_in[14];
  const float* logits_b   = (const float*)d_in[15];

  // ---- workspace layout ----
  char*  ws  = (char*)d_ws;
  size_t szh = (size_t)M * DIM * sizeof(float);                 // 64 MB
  float*    h     = (float*)(ws);
  float*    hn    = (float*)(ws + szh);
  _Float16* hn16  = (_Float16*)(ws + 2 * szh);                  // M*D f16
  _Float16* qkv16 = (_Float16*)(ws + 2 * szh + szh / 2);        // M*3D f16
  _Float16* o16   = (_Float16*)(ws + 4 * szh);                  // M*D f16
  _Float16* f16b  = (_Float16*)(ws + 4 * szh + szh / 2);        // M*FF f16
  _Float16* w16   = (_Float16*)(ws + 5 * szh + szh / 4);        // weights f16
  _Float16* wqkv16 = w16;
  _Float16* wpo16  = wqkv16 + (size_t)NL * 3 * DIM * DIM;
  _Float16* wl116  = wpo16 + (size_t)NL * DIM * DIM;
  _Float16* wl216  = wl116 + (size_t)NL * FFD * DIM;

  // ---- stage all weights to f16 once ----
  {
    long n1 = (long)NL * 3 * DIM * DIM;
    long n2 = (long)NL * DIM * DIM;
    long n3 = (long)NL * FFD * DIM;
    long n4 = (long)NL * DIM * FFD;
    cvt16_kernel<<<(unsigned)((n1 + 255) / 256), 256, 0, stream>>>(proj_in_w,  wqkv16, n1);
    cvt16_kernel<<<(unsigned)((n2 + 255) / 256), 256, 0, stream>>>(proj_out_w, wpo16,  n2);
    cvt16_kernel<<<(unsigned)((n3 + 255) / 256), 256, 0, stream>>>(lin1_w,     wl116,  n3);
    cvt16_kernel<<<(unsigned)((n4 + 255) / 256), 256, 0, stream>>>(lin2_w,     wl216,  n4);
  }

  long total = (long)M * DIM;
  embed_kernel<<<(unsigned)((total + 255) / 256), 256, 0, stream>>>(x, embed_w, h, total);

  for (int l = 0; l < NL; ++l) {
    // hn, hn16 = LN(h, norm1)
    layernorm_kernel<<<(M + 7) / 8, 256, 0, stream>>>(
        h, norm1_w + l * DIM, norm1_b + l * DIM, hn, hn16, M);
    // qkv16 = hn16 * Win^T (f16 out)
    int tiles = (M / 32) * (3 * DIM / 64);
    gemm16_kernel<0, 0, 0, 1><<<(tiles + 7) / 8, 256, 0, stream>>>(
        hn16, wqkv16 + (size_t)l * 3 * DIM * DIM, nullptr, nullptr,
        nullptr, qkv16, M, 3 * DIM, DIM);
    // attention -> o16
    attn_kernel<<<Bn * NH, 256, 0, stream>>>(qkv16, att, o16);
    // h = o16 * Wout^T + bout + hn
    tiles = (M / 32) * (DIM / 64);
    gemm16_kernel<1, 0, 1, 0><<<(tiles + 7) / 8, 256, 0, stream>>>(
        o16, wpo16 + (size_t)l * DIM * DIM, proj_out_b + l * DIM, hn,
        h, nullptr, M, DIM, DIM);
    // hn, hn16 = LN(h, norm2)
    layernorm_kernel<<<(M + 7) / 8, 256, 0, stream>>>(
        h, norm2_w + l * DIM, norm2_b + l * DIM, hn, hn16, M);
    // f16b = gelu(hn16 * W1^T + b1) (f16 out)
    tiles = (M / 32) * (FFD / 64);
    gemm16_kernel<1, 1, 0, 1><<<(tiles + 7) / 8, 256, 0, stream>>>(
        hn16, wl116 + (size_t)l * FFD * DIM, lin1_b + l * FFD, nullptr,
        nullptr, f16b, M, FFD, DIM);
    // h = f16b * W2^T + b2 + hn
    tiles = (M / 32) * (DIM / 64);
    gemm16_kernel<1, 0, 1, 0><<<(tiles + 7) / 8, 256, 0, stream>>>(
        f16b, wl216 + (size_t)l * DIM * FFD, lin2_b + l * DIM, hn,
        h, nullptr, M, DIM, FFD);
  }

  logits_kernel<<<Bn, 256, 0, stream>>>(h, logits_w, logits_b, (float*)d_out);
}
